// ContrastiveLoss_3332894621778
// MI455X (gfx1250) — matrix operand order repaired and checked by reading, not compiled, not run on hardware
//
#include <hip/hip_runtime.h>
#include <hip/hip_bf16.h>
#include <math.h>

// ---------------------------------------------------------------------------
// ContrastiveLoss (SimCLR-style CE over cosine-similarity logits), fused.
//   N = 4096 rows, D = 256 features, T = 0.5, labels = arange(N)
//   loss = mean_r [ logsumexp_c(2 * <zi_r, zj_c>) - 2 * <zi_r, zj_r> ]
// Rows are unit-normalized -> |logit| <= 2 -> exp() cannot overflow, so the
// softmax needs no running max. The diagonal term is recomputed directly in
// the reduction kernel (1M MACs total) to keep the WMMA loop branch-free.
// ---------------------------------------------------------------------------

typedef _Float16 v16h __attribute__((ext_vector_type(16)));
typedef _Float16 h8   __attribute__((ext_vector_type(8)));
typedef float    v8f  __attribute__((ext_vector_type(8)));

#define N_ROWS 4096
#define D_FEAT 256
#define N_TILES (N_ROWS / 16)
#define INV_T  2.0f        // 1 / temperature (temperature = 0.5)
#define NEPS   1e-12f

// -------------------------------------------------------------------
// Kernel 1: L2-normalize each row (f32 in) and emit f16 row-major out.
// blockIdx.x in [0, 2N): first N blocks -> z_i, next N -> z_j.
// -------------------------------------------------------------------
__global__ void normalize_f16_kernel(const float* __restrict__ zi,
                                     const float* __restrict__ zj,
                                     _Float16* __restrict__ zi16,
                                     _Float16* __restrict__ zj16) {
    const int b    = blockIdx.x;
    const int row  = b & (N_ROWS - 1);
    const bool isI = (b < N_ROWS);
    const float* __restrict__ src = isI ? zi : zj;
    _Float16* __restrict__ dst    = isI ? zi16 : zj16;

    const int tid  = threadIdx.x;
    const int lane = tid & 31;
    const int wv   = tid >> 5;

    const float v = src[(size_t)row * D_FEAT + tid];
    float ss = v * v;
    #pragma unroll
    for (int m = 16; m >= 1; m >>= 1)
        ss += __shfl_xor(ss, m, 32);

    __shared__ float part[8];
    __shared__ float s_scale;
    if (lane == 0) part[wv] = ss;
    __syncthreads();
    if (tid == 0) {
        float tot = 0.0f;
        #pragma unroll
        for (int w = 0; w < 8; ++w) tot += part[w];
        s_scale = 1.0f / fmaxf(sqrtf(tot), NEPS);
    }
    __syncthreads();
    dst[(size_t)row * D_FEAT + tid] = (_Float16)(v * s_scale);
}

// -------------------------------------------------------------------
// Helpers for kernel 2.
// -------------------------------------------------------------------
__device__ __forceinline__ void load_btile(const _Float16* __restrict__ base,
                                           int bsel, v16h* buf) {
    // B layout (32x16 f16): lanes 0-15 hold N=lane, K=0..15; lanes 16-31 hold
    // N=lane-16, K=16..31 -> one contiguous 32-byte load per fragment.
    #pragma unroll
    for (int kk = 0; kk < 8; ++kk)
        buf[kk] = *(const v16h*)(base + 32 * kk + bsel);
}

__device__ __forceinline__ void mma_tile(const v16h* afrag, const v16h* buf,
                                         float* esum) {
    v8f c = {0.0f, 0.0f, 0.0f, 0.0f, 0.0f, 0.0f, 0.0f, 0.0f};
    #pragma unroll
    for (int kk = 0; kk < 8; ++kk)
        c = __builtin_amdgcn_wmma_f32_16x16x32_f16(
                /*neg_a=*/false, afrag[kk],
                /*neg_b=*/false, buf[kk],
                /*c_mod=*/(short)0, c,
                /*reuse_a=*/false, /*reuse_b=*/false);
    // fold exp(logit) into per-lane row sums (logit = 2 * dot); trans-op
    // latency overlaps the already-issued next-tile loads.
    #pragma unroll
    for (int r = 0; r < 8; ++r)
        esum[r] += __expf(INV_T * c[r]);
}

// -------------------------------------------------------------------
// Kernel 2: fused GEMM (WMMA f16 -> f32 accumulate) + row logsumexp.
// One block = 16 output rows; 8 waves each sweep 32 column tiles with
// full tile-level double buffering of the B fragments (ping-pong).
// -------------------------------------------------------------------
__global__ void
__launch_bounds__(256, 2)
fused_gemm_lse_kernel(const _Float16* __restrict__ zi16,
                      const _Float16* __restrict__ zj16,
                      float* __restrict__ rowLse) {
    const int blockRow = blockIdx.x;             // 0..255 (row tile index)
    const int tid  = threadIdx.x;
    const int lane = tid & 31;
    const int wv   = tid >> 5;                   // 0..7
    const int l15  = lane & 15;
    const bool hi16 = (lane >= 16);
    const int bsel  = hi16 ? 16 : 0;             // B lane K-half select

    __shared__ float s_wsum[8][16];

    // ---- Preload the 8 A-fragments for this 16-row tile (K = 0..255) ----
    // A layout used here (16x32): lanes 0-15 hold M=lane, K=0..15;
    // lanes 16-31 hold M=lane-16, K=16..31 (matches B's contiguous K order).
    const _Float16* __restrict__ Arow =
        zi16 + (size_t)(blockRow * 16 + l15) * D_FEAT;
    v16h afrag[8];
    #pragma unroll
    for (int kk = 0; kk < 8; ++kk) {
        const int koff = 32 * kk + (hi16 ? 16 : 0);
        h8 lo  = *(const h8*)(Arow + koff);
        h8 hiv = *(const h8*)(Arow + koff + 8);
        afrag[kk] = __builtin_shufflevector(lo, hiv,
                        0,1,2,3,4,5,6,7, 8,9,10,11,12,13,14,15);
    }

    float esum[8];
    #pragma unroll
    for (int r = 0; r < 8; ++r) esum[r] = 0.0f;

    // B row base for column tile index j
    #define BROW(j) (zj16 + (size_t)((j) * 16 + l15) * D_FEAT)

    // ---- Double-buffered sweep: unroll by 2, ping-pong bufA/bufB ----
    v16h bufA[8], bufB[8];
    load_btile(BROW(wv), bsel, bufA);            // first tile, buffer A

    // jt = wv, wv+16, ...: processes tiles jt (bufA) and jt+8 (bufB);
    // with wv <= 7 both are always < N_TILES, so the body is branch-free.
    for (int jt = wv; jt < N_TILES; jt += 16) {
        load_btile(BROW(jt + 8), bsel, bufB);    // prefetch tile jt+8
        mma_tile(afrag, bufA, esum);             // compute tile jt

        const int jn = (jt + 16 < N_TILES) ? (jt + 16) : wv;  // wrap: harmless
        load_btile(BROW(jn), bsel, bufA);        // prefetch tile jt+16
        mma_tile(afrag, bufB, esum);             // compute tile jt+8
    }
    #undef BROW

    // ---- Combine 16 column-lanes within each half (rows 0-7 / 8-15) ----
    #pragma unroll
    for (int r = 0; r < 8; ++r) {
        #pragma unroll
        for (int m = 1; m <= 8; m <<= 1)
            esum[r] += __shfl_xor(esum[r], m, 16);
    }
    if (lane == 0) {
        #pragma unroll
        for (int r = 0; r < 8; ++r) s_wsum[wv][r] = esum[r];
    }
    if (lane == 16) {
        #pragma unroll
        for (int r = 0; r < 8; ++r) s_wsum[wv][r + 8] = esum[r];
    }
    __syncthreads();

    // ---- One thread per row: combine 8 waves, emit row logsumexp ----
    if (tid < 16) {
        float tot = 0.0f;
        #pragma unroll
        for (int w = 0; w < 8; ++w) tot += s_wsum[w][tid];
        rowLse[blockRow * 16 + tid] = __logf(tot);   // max-free: |logit| <= 2
    }
}

// -------------------------------------------------------------------
// Kernel 3: recompute diagonal dots + deterministic fixed-order mean.
// 256 threads, 16 rows/thread; all data L2-resident (4 MB).
// -------------------------------------------------------------------
__global__ void final_reduce_kernel(const _Float16* __restrict__ zi16,
                                    const _Float16* __restrict__ zj16,
                                    const float* __restrict__ rowLse,
                                    float* __restrict__ out) {
    __shared__ float s[256];
    const int tid = threadIdx.x;
    float acc = 0.0f;
    #pragma unroll
    for (int i = 0; i < N_ROWS / 256; ++i) {
        const int row = tid + i * 256;
        const _Float16* __restrict__ a = zi16 + (size_t)row * D_FEAT;
        const _Float16* __restrict__ b = zj16 + (size_t)row * D_FEAT;
        float dot = 0.0f;
        #pragma unroll 4
        for (int k = 0; k < D_FEAT / 8; ++k) {
            h8 av = *(const h8*)(a + k * 8);
            h8 bv = *(const h8*)(b + k * 8);
            #pragma unroll
            for (int e = 0; e < 8; ++e)
                dot = fmaf((float)av[e], (float)bv[e], dot);
        }
        acc += rowLse[row] - INV_T * dot;
    }
    s[tid] = acc;
    __syncthreads();
    #pragma unroll
    for (int off = 128; off >= 1; off >>= 1) {
        if (tid < off) s[tid] += s[tid + off];
        __syncthreads();
    }
    if (tid == 0) out[0] = s[0] * (1.0f / (float)N_ROWS);
}

// -------------------------------------------------------------------
extern "C" void kernel_launch(void* const* d_in, const int* in_sizes, int n_in,
                              void* d_out, int out_size, void* d_ws, size_t ws_size,
                              hipStream_t stream) {
    const float* zi = (const float*)d_in[0];
    const float* zj = (const float*)d_in[1];
    float* out = (float*)d_out;

    // workspace layout: [zi16: 2MB][zj16: 2MB][rowLse: 16KB]
    char* wsb = (char*)d_ws;
    _Float16* zi16 = (_Float16*)wsb;
    _Float16* zj16 = zi16 + (size_t)N_ROWS * D_FEAT;
    float* rowLse  = (float*)(wsb + 2 * (size_t)N_ROWS * D_FEAT * sizeof(_Float16));

    normalize_f16_kernel<<<2 * N_ROWS, 256, 0, stream>>>(zi, zj, zi16, zj16);
    fused_gemm_lse_kernel<<<N_TILES, 256, 0, stream>>>(zi16, zj16, rowLse);
    final_reduce_kernel<<<1, 256, 0, stream>>>(zi16, zj16, rowLse, out);
}
